// WindowedAttention3D_63196148793673
// MI455X (gfx1250) — matrix operand order repaired and checked by reading, not compiled, not run on hardware
//
#include <hip/hip_runtime.h>

typedef __attribute__((ext_vector_type(16))) _Float16 v16h;
typedef __attribute__((ext_vector_type(8)))  _Float16 v8h;
typedef __attribute__((ext_vector_type(8)))  float    v8f;
typedef __attribute__((ext_vector_type(4)))  float    v4f;

// ---------------- WMMA helper ----------------
__device__ __forceinline__ v8f wmma16(v16h a, v16h b, v8f c) {
  return __builtin_amdgcn_wmma_f32_16x16x32_f16(false, a, false, b, (short)0, c, false, false);
}

// A operand (16x32) from row-major f16 (stride in halves, rows 16B-aligned): 2x ds_load_b128
__device__ __forceinline__ v16h load_a16(const _Float16* base, int stride, int lane) {
  const int m = lane & 15;
  const int koff = (lane & 16) ? 16 : 0;
  const _Float16* rp = base + m * stride + koff;
  v8h lo = *(const v8h*)(rp);
  v8h hi = *(const v8h*)(rp + 8);
  v16h a;
#pragma unroll
  for (int i = 0; i < 8; ++i) { a[i] = lo[i]; a[8 + i] = hi[i]; }
  return a;
}

// B operand (32x16) from B^T row-major (N x K): B[k][n] = src[n*stride + k]: 2x ds_load_b128
__device__ __forceinline__ v16h load_bt16(const _Float16* base, int stride, int lane) {
  const int n = lane & 15;
  const int kb = (lane & 16) ? 16 : 0;
  const _Float16* rp = base + n * stride + kb;
  v8h lo = *(const v8h*)(rp);
  v8h hi = *(const v8h*)(rp + 8);
  v16h b;
#pragma unroll
  for (int i = 0; i < 8; ++i) { b[i] = lo[i]; b[8 + i] = hi[i]; }
  return b;
}

__device__ __forceinline__ float redmax16(float x) {
  x = fmaxf(x, __shfl_xor(x, 1, 32));
  x = fmaxf(x, __shfl_xor(x, 2, 32));
  x = fmaxf(x, __shfl_xor(x, 4, 32));
  x = fmaxf(x, __shfl_xor(x, 8, 32));
  return x;
}
__device__ __forceinline__ float redsum16(float x) {
  x += __shfl_xor(x, 1, 32);
  x += __shfl_xor(x, 2, 32);
  x += __shfl_xor(x, 4, 32);
  x += __shfl_xor(x, 8, 32);
  return x;
}
__device__ __forceinline__ float redmax8(float x) {
  x = fmaxf(x, __shfl_xor(x, 1, 32));
  x = fmaxf(x, __shfl_xor(x, 2, 32));
  x = fmaxf(x, __shfl_xor(x, 4, 32));
  return x;
}
__device__ __forceinline__ float redsum8(float x) {
  x += __shfl_xor(x, 1, 32);
  x += __shfl_xor(x, 2, 32);
  x += __shfl_xor(x, 4, 32);
  return x;
}

// ---------------- prep1: weight pre-swizzle (B-operand layout) + kg/vg gemm ----------------
__global__ __launch_bounds__(256) void prep_kernel(
    const float* __restrict__ Wqkv, const float* __restrict__ Wqg,
    const float* __restrict__ Wout, const float* __restrict__ Woutg,
    const float* __restrict__ Wkvg, const float* __restrict__ bkvg,
    const float* __restrict__ gtok,
    _Float16* __restrict__ W1sw, _Float16* __restrict__ Woutsw,
    _Float16* __restrict__ Woutgsw, float* __restrict__ kvg) {
  const int e = blockIdx.x * 256 + threadIdx.x;
  const int NW1 = 65536, NWO = 16384;
  if (e < NW1) {
    int idx = e & 15, lane = (e >> 4) & 31, tile = e >> 9;
    int nt = tile & 31, kt = tile >> 5;  // NT = 32
    int n = nt * 16 + (lane & 15);
    int k = kt * 32 + ((lane & 16) ? 16 : 0) + idx;
    W1sw[e] = (_Float16)((n < 384) ? Wqkv[k * 384 + n] : Wqg[k * 128 + (n - 384)]);
  } else if (e < NW1 + 2 * NWO) {
    int f = (e - NW1) & (NWO - 1);
    bool isG = (e - NW1) >= NWO;
    int idx = f & 15, lane = (f >> 4) & 31, tile = f >> 9;
    int nt = tile & 7, kt = tile >> 3;  // NT = 8
    int n = nt * 16 + (lane & 15);
    int k = kt * 32 + ((lane & 16) ? 16 : 0) + idx;
    if (!isG) Woutsw[f]  = (_Float16)Wout[k * 128 + n];
    else      Woutgsw[f] = (_Float16)Woutg[k * 128 + n];
  } else if (e < NW1 + 2 * NWO + 2048) {
    int f = e - NW1 - 2 * NWO;          // kvg = g_tokens(8x128) @ Wkvg(128x256) + bkvg
    int g = f >> 8, col = f & 255;
    float s = bkvg[col];
    for (int k0 = 0; k0 < 128; ++k0) s += gtok[g * 128 + k0] * Wkvg[k0 * 256 + col];
    kvg[f] = s;
  }
}

// ---------------- prep2: pack kg^T / 0.1*vg as zero-padded WMMA B tiles (per head) --------
__global__ __launch_bounds__(256) void prep2_kernel(
    const float* __restrict__ kvg, _Float16* __restrict__ kgsw, _Float16* __restrict__ vgsw) {
  const int e = blockIdx.x * 256 + threadIdx.x;  // 0..8191
  if (e >= 8192) return;
  int f = e & 4095;
  int idx = f & 15, lane = (f >> 4) & 31, h = f >> 9;
  int n = lane & 15;
  int k = ((lane & 16) ? 16 : 0) + idx;
  if (e < 4096) {  // kg^T: B[k=d][n=gtoken]
    float val = (k < 16 && n < 8) ? kvg[n * 256 + 16 * h + k] : 0.f;
    kgsw[f] = (_Float16)val;
  } else {         // 0.1*vg: B[k=gtoken][n=d]
    float val = (k < 8) ? 0.1f * kvg[k * 256 + 128 + 16 * h + n] : 0.f;
    vgsw[f] = (_Float16)val;
  }
}

// ---------------- main: one 4x4x4 window per block, 8 waves, one head per wave ----------------
#define XS 136   // 64 x 128 row-major buffers (Xbuf/Obuf alias, Gbuf)
#define QS 40    // Qpad/Kpad/QGpad: per-head 64 x 32 (cols 16..31 stored zeros)
#define VS 72    // Vt: per-head 16 x 64 (transposed V)
#define PS 40    // P staging per wave: 16 x 32

#define OFF_X    0                 // 17408 (aliased: Obuf after GEMM1)
#define OFF_Q    17408             // 8*64*40*2 = 40960
#define OFF_K    58368             // 40960
#define OFF_V    99328             // 8*16*72*2 = 18432
#define OFF_QG   117760            // 40960 (qg in padded A layout)
#define OFF_P    158720            // 8*16*40*2 = 10240
#define OFF_G    168960            // 17408
#define SMEM_BYTES 186368

__global__ __launch_bounds__(256) void win_attn_kernel(
    const float* __restrict__ feat,
    const v16h* __restrict__ W1sw, const v16h* __restrict__ Woutsw,
    const v16h* __restrict__ Woutgsw,
    const v16h* __restrict__ kgsw, const v16h* __restrict__ vgsw,
    const float* __restrict__ bqkv, const float* __restrict__ bqg,
    const float* __restrict__ bout, const float* __restrict__ boutg,
    float* __restrict__ outp) {
  extern __shared__ char smem[];
  _Float16* Xbuf  = (_Float16*)(smem + OFF_X);
  _Float16* Obuf  = (_Float16*)(smem + OFF_X);   // alias: X dead after GEMM1 barrier
  _Float16* Qpad  = (_Float16*)(smem + OFF_Q);
  _Float16* Kpad  = (_Float16*)(smem + OFF_K);
  _Float16* Vt    = (_Float16*)(smem + OFF_V);
  _Float16* QGpad = (_Float16*)(smem + OFF_QG);
  _Float16* Pstg  = (_Float16*)(smem + OFF_P);
  _Float16* Gbuf  = (_Float16*)(smem + OFF_G);

  const int tid = threadIdx.x;
  const int lane = tid & 31;
  const int w = tid >> 5;
  const int bid = blockIdx.x;
  const int batch = bid >> 9;
  const int widx = bid & 511;
  const int wd = widx >> 6, wh = (widx >> 3) & 7, ww = widx & 7;
  const int hi8 = (lane & 16) ? 8 : 0;
  const int n16 = lane & 15;

  // ---- stage window X (float4 gather -> packed f16 b128 stores); zero pad columns ----
  {
    int t = tid >> 2;
    int c0 = (tid & 3) * 32;
    int z = t >> 4, y = (t >> 2) & 3, x = t & 3;
    size_t gbase = ((((size_t)(batch * 32 + wd * 4 + z) * 32) + (wh * 4 + y)) * 32 + (ww * 4 + x)) * 128;
    const v4f* fp = (const v4f*)(feat + gbase + c0);
#pragma unroll
    for (int qq = 0; qq < 4; ++qq) {
      v4f f0 = fp[2 * qq], f1 = fp[2 * qq + 1];
      v8h hv;
#pragma unroll
      for (int i = 0; i < 4; ++i) { hv[i] = (_Float16)f0[i]; hv[4 + i] = (_Float16)f1[i]; }
      *(v8h*)(Xbuf + t * XS + c0 + qq * 8) = hv;
    }
    v8h z8;
#pragma unroll
    for (int i = 0; i < 8; ++i) z8[i] = (_Float16)0.0f;
    // zero d=16..31 of Qpad/Kpad/QGpad: 1536 rows x 32B, 6 rows per thread
#pragma unroll
    for (int j = 0; j < 6; ++j) {
      int r = tid * 6 + j;
      _Float16* bp = ((r < 512) ? Qpad : (r < 1024) ? Kpad : QGpad) + (r & 511) * QS + 16;
      *(v8h*)bp = z8;
      *(v8h*)(bp + 8) = z8;
    }
  }
  __syncthreads();

  // ---- GEMM1: [Q|K|V|qg] = X(64x128) @ W1(128x512).
  //      Wave w owns nt = 4w..4w+3 (one dest region per wave), all 4 mt:
  //      every B tile loaded exactly once per block. ----
  {
    const int reg = __builtin_amdgcn_readfirstlane(w >> 1);  // 0=Q 1=K 2=V 3=qg
    v16h acache[4][4];
#pragma unroll
    for (int mt = 0; mt < 4; ++mt)
#pragma unroll
      for (int kt = 0; kt < 4; ++kt)
        acache[mt][kt] = load_a16(Xbuf + mt * 16 * XS + kt * 32, XS, lane);
#pragma unroll
    for (int j = 0; j < 4; ++j) {
      const int nt = w * 4 + j;
      v16h b0 = W1sw[(0 * 32 + nt) * 32 + lane];
      v16h b1 = W1sw[(1 * 32 + nt) * 32 + lane];
      v16h b2 = W1sw[(2 * 32 + nt) * 32 + lane];
      v16h b3 = W1sw[(3 * 32 + nt) * 32 + lane];
      v8f acc[4];
#pragma unroll
      for (int mt = 0; mt < 4; ++mt) {
        v8f a = {0, 0, 0, 0, 0, 0, 0, 0};
        a = wmma16(acache[mt][0], b0, a);
        a = wmma16(acache[mt][1], b1, a);
        a = wmma16(acache[mt][2], b2, a);
        a = wmma16(acache[mt][3], b3, a);
        acc[mt] = a;
      }
      const int col = nt * 16 + n16;
      if (reg == 0) {          // Q: bias + 0.25 score scale
        int h = (w * 4 + j) & 7;  // == nt since nt<8
        float bias = bqkv[col];
#pragma unroll
        for (int mt = 0; mt < 4; ++mt)
#pragma unroll
          for (int v = 0; v < 8; ++v)
            Qpad[h * 64 * QS + (mt * 16 + hi8 + v) * QS + n16] = (_Float16)((acc[mt][v] + bias) * 0.25f);
      } else if (reg == 1) {   // K
        int h = (nt - 8) & 7;
        float bias = bqkv[col];
#pragma unroll
        for (int mt = 0; mt < 4; ++mt)
#pragma unroll
          for (int v = 0; v < 8; ++v)
            Kpad[h * 64 * QS + (mt * 16 + hi8 + v) * QS + n16] = (_Float16)(acc[mt][v] + bias);
      } else if (reg == 2) {   // V, transposed: Vt[h][d][token]
        int h = (nt - 16) & 7;
        float bias = bqkv[col];
#pragma unroll
        for (int mt = 0; mt < 4; ++mt)
#pragma unroll
          for (int v = 0; v < 8; ++v)
            Vt[h * 16 * VS + n16 * VS + (mt * 16 + hi8 + v)] = (_Float16)(acc[mt][v] + bias);
      } else {                 // qg: padded A layout, bias + 0.25 scale folded
        int h = (nt - 24) & 7;
        float bias = bqg[col - 384];
#pragma unroll
        for (int mt = 0; mt < 4; ++mt)
#pragma unroll
          for (int v = 0; v < 8; ++v)
            QGpad[h * 64 * QS + (mt * 16 + hi8 + v) * QS + n16] = (_Float16)((acc[mt][v] + bias) * 0.25f);
      }
    }
  }
  __syncthreads();

  // ---- attention: one head per wave ----
  {
    const int h = w;
    const _Float16* Qh  = Qpad + h * 64 * QS;
    const _Float16* Kh  = Kpad + h * 64 * QS;
    const _Float16* Vth = Vt + h * 16 * VS;
    _Float16* Pw = Pstg + h * 16 * PS;

    // local windowed attention, flash softmax over 32-wide chunks
    for (int mt = 0; mt < 4; ++mt) {
      v16h qa = load_a16(Qh + mt * 16 * QS, QS, lane);   // K 16..31 stored zeros
      float mrow[8], lsum[8];
      v8f o = {0, 0, 0, 0, 0, 0, 0, 0};
#pragma unroll
      for (int v = 0; v < 8; ++v) { mrow[v] = -3.0e38f; lsum[v] = 0.0f; }
#pragma unroll
      for (int ntp = 0; ntp < 2; ++ntp) {
        v16h kb0 = load_bt16(Kh + (ntp * 2 + 0) * 16 * QS, QS, lane);
        v16h kb1 = load_bt16(Kh + (ntp * 2 + 1) * 16 * QS, QS, lane);
        v8f z = {0, 0, 0, 0, 0, 0, 0, 0};
        v8f s0 = wmma16(qa, kb0, z);
        v8f s1 = wmma16(qa, kb1, z);
#pragma unroll
        for (int v = 0; v < 8; ++v) {
          float tm = fmaxf(redmax16(s0[v]), redmax16(s1[v]));
          float nm = fmaxf(mrow[v], tm);
          float cf = __expf(mrow[v] - nm);
          float p0 = __expf(s0[v] - nm);
          float p1 = __expf(s1[v] - nm);
          lsum[v] = lsum[v] * cf + redsum16(p0) + redsum16(p1);
          mrow[v] = nm;
          o[v] *= cf;
          Pw[(v + hi8) * PS + n16]      = (_Float16)p0;
          Pw[(v + hi8) * PS + 16 + n16] = (_Float16)p1;
        }
        v16h pa = load_a16(Pw, PS, lane);
        v16h vb = load_bt16(Vth + ntp * 32, VS, lane);
        o = wmma16(pa, vb, o);
      }
#pragma unroll
      for (int v = 0; v < 8; ++v)
        Obuf[(mt * 16 + v + hi8) * XS + 16 * h + n16] = (_Float16)(o[v] / lsum[v]);
    }

    // global-token cross attention via WMMA (kg^T / 0.1*vg pre-padded B tiles)
    {
      v16h kgB = kgsw[h * 32 + lane];
      v16h vgB = vgsw[h * 32 + lane];
      const _Float16* QGh = QGpad + h * 64 * QS;
      for (int mt = 0; mt < 4; ++mt) {
        v16h qa = load_a16(QGh + mt * 16 * QS, QS, lane);
        v8f z = {0, 0, 0, 0, 0, 0, 0, 0};
        v8f sg = wmma16(qa, kgB, z);   // cols 8..15 are zeros (pad)
#pragma unroll
        for (int v = 0; v < 8; ++v) {
          float mx = redmax8(sg[v]);           // reduce over the 8 real cols' lane group
          float ev = __expf(sg[v] - mx);
          float ssum = redsum8(ev);
          float pv = (n16 < 8) ? (ev / ssum) : 0.f;
          Pw[(v + hi8) * PS + n16]      = (_Float16)pv;
          Pw[(v + hi8) * PS + 16 + n16] = (_Float16)0.0f;
        }
        v16h pa = load_a16(Pw, PS, lane);
        v8f og = wmma16(pa, vgB, z);
#pragma unroll
        for (int v = 0; v < 8; ++v)
          Gbuf[(mt * 16 + v + hi8) * XS + 16 * h + n16] = (_Float16)og[v];
      }
    }
  }
  __syncthreads();

  // ---- fused epilogue: out = O@Wout + (0.1*G)@Woutg + bout + 0.1*boutg.
  //      Wave w owns nt = w: each Wout/Woutg tile loaded exactly once per block. ----
  {
    v8f acc[4];
#pragma unroll
    for (int mt = 0; mt < 4; ++mt) acc[mt] = (v8f){0, 0, 0, 0, 0, 0, 0, 0};
#pragma unroll
    for (int kt = 0; kt < 4; ++kt) {
      v16h b = Woutsw[(kt * 8 + w) * 32 + lane];
#pragma unroll
      for (int mt = 0; mt < 4; ++mt)
        acc[mt] = wmma16(load_a16(Obuf + mt * 16 * XS + kt * 32, XS, lane), b, acc[mt]);
    }
#pragma unroll
    for (int kt = 0; kt < 4; ++kt) {
      v16h b = Woutgsw[(kt * 8 + w) * 32 + lane];
#pragma unroll
      for (int mt = 0; mt < 4; ++mt)
        acc[mt] = wmma16(load_a16(Gbuf + mt * 16 * XS + kt * 32, XS, lane), b, acc[mt]);
    }
    const int col = w * 16 + n16;
    const float bias = bout[col] + 0.1f * boutg[col];
#pragma unroll
    for (int mt = 0; mt < 4; ++mt)
#pragma unroll
      for (int v = 0; v < 8; ++v) {
        int t = mt * 16 + v + hi8;
        int z = t >> 4, y = (t >> 2) & 3, x = t & 3;
        size_t gi = ((((size_t)(batch * 32 + wd * 4 + z) * 32) + (wh * 4 + y)) * 32 + (ww * 4 + x)) * 128 + col;
        outp[gi] = acc[mt][v] + bias;
      }
  }
}

extern "C" void kernel_launch(void* const* d_in, const int* in_sizes, int n_in,
                              void* d_out, int out_size, void* d_ws, size_t ws_size,
                              hipStream_t stream) {
  const float* feat  = (const float*)d_in[0];
  const float* Wqkv  = (const float*)d_in[1];
  const float* bqkv  = (const float*)d_in[2];
  const float* Wout  = (const float*)d_in[3];
  const float* bout  = (const float*)d_in[4];
  const float* Wqg   = (const float*)d_in[5];
  const float* bqg   = (const float*)d_in[6];
  const float* Wkvg  = (const float*)d_in[7];
  const float* bkvg  = (const float*)d_in[8];
  const float* Woutg = (const float*)d_in[9];
  const float* boutg = (const float*)d_in[10];
  const float* gtok  = (const float*)d_in[11];
  float* outp = (float*)d_out;

  char* ws = (char*)d_ws;
  _Float16* W1sw    = (_Float16*)ws;                       // 131072 B
  _Float16* Woutsw  = (_Float16*)(ws + 131072);            // 32768 B
  _Float16* Woutgsw = (_Float16*)(ws + 163840);            // 32768 B
  float*    kvgws   = (float*)(ws + 196608);               // 8192 B
  _Float16* kgsw    = (_Float16*)(ws + 204800);            // 8192 B
  _Float16* vgsw    = (_Float16*)(ws + 212992);            // 8192 B

  (void)in_sizes; (void)n_in; (void)out_size; (void)ws_size;

  hipFuncSetAttribute((const void*)win_attn_kernel,
                      hipFuncAttributeMaxDynamicSharedMemorySize, SMEM_BYTES);

  prep_kernel<<<392, 256, 0, stream>>>(Wqkv, Wqg, Wout, Woutg, Wkvg, bkvg, gtok,
                                       W1sw, Woutsw, Woutgsw, kvgws);
  prep2_kernel<<<32, 256, 0, stream>>>(kvgws, kgsw, vgsw);
  win_attn_kernel<<<2048, 256, SMEM_BYTES, stream>>>(
      feat, (const v16h*)W1sw, (const v16h*)Woutsw, (const v16h*)Woutgsw,
      (const v16h*)kgsw, (const v16h*)vgsw,
      bqkv, bqg, bout, boutg, outp);
}
